// SchNet_66039417143595
// MI455X (gfx1250) — compile-verified, matrix operand order checked by reading, not run on hardware
//
#include <hip/hip_runtime.h>

#define XDIM 32
#define HIDC 128
#define NFC  128
#define NGC  50
#define KP   64     // NG padded to WMMA K multiple
#define KPP  72     // padded LDS pitch for 64-wide tiles (bank spread, 16B aligned)
#define HPP  136    // padded LDS pitch for 128-wide tiles (bank spread, 16B aligned)
#define NLC  6
#define SHIDC 64
#define CUTOFF 10.0f

typedef __attribute__((ext_vector_type(16))) _Float16 v16h;
typedef __attribute__((ext_vector_type(8)))  _Float16 v8h;
typedef __attribute__((ext_vector_type(8)))  float    v8f;

// ---------------- WMMA helpers (CDNA5 16-bit layouts, wave32) ----------------

__device__ __forceinline__ v8f wmma_f16(v16h a, v16h b, v8f c) {
  return __builtin_amdgcn_wmma_f32_16x16x32_f16(false, a, false, b, (short)0, c, false, false);
}

__device__ __forceinline__ v16h make16(v8h lo, v8h hi) {
  return __builtin_shufflevector(lo, hi, 0, 1, 2, 3, 4, 5, 6, 7, 8, 9, 10, 11, 12, 13, 14, 15);
}

// A: 16xK f16 tile, row-major in LDS (pitch halves, rows 16B-aligned); kc = K base.
// Lane L<16 holds row M=L with K {0..7, 16..23}; lane L>=16 row M=L-16, K {8..15, 24..31}.
// Two contiguous 8-half runs -> two aligned ds_load_b128.
__device__ __forceinline__ v16h load_frag_a(const _Float16* base, int pitch, int kc, int lane) {
  int row = lane & 15;
  int hf  = lane >> 4;
  const _Float16* p = base + row * pitch + kc + hf * 8;
  v8h lo = *(const v8h*)p;
  v8h hi = *(const v8h*)(p + 16);
  return make16(lo, hi);
}

// B fragments pre-swizzled into two half-regions per 512-half tile:
//   lo j=0..7  at  lane*8 + j
//   hi j=8..15 at  256 + lane*8 + (j-8)
// -> per ds_load_b128, lane stride is 16B: full 64-bank coverage (2-cycle minimum).
__device__ __forceinline__ v16h load_frag_b_swz(const _Float16* sW, int KC, int nt, int chunk, int lane) {
  const _Float16* p = sW + (((nt * KC + chunk) << 9)) + lane * 8;
  v8h lo = *(const v8h*)p;
  v8h hi = *(const v8h*)(p + 256);
  return make16(lo, hi);
}

// swizzled index of W[k][n] (K x 128 row-major source) in fragment-major layout
__device__ __forceinline__ int swz_idx(int k, int n, int KC) {
  int chunk = k >> 5, kk = k & 31;
  int nt = n >> 4, lx = n & 15;
  int hf = (kk >> 3) & 1;
  int j  = (kk & 7) + 8 * ((kk >> 4) & 1);
  int lane = lx + 16 * hf;
  int within = (j < 8) ? (lane * 8 + j) : (256 + lane * 8 + (j - 8));
  return (nt * KC + chunk) * 512 + within;
}

// ---------------- init / weight conversion ----------------

__global__ void zero_ws_k(float* __restrict__ agg, float* __restrict__ stats, long long n) {
  long long i = (long long)blockIdx.x * 256 + threadIdx.x;
  if (i < n)    agg[i]   = 0.f;
  if (i < 1024) stats[i] = 0.f;
}

__global__ void convert_weights_k(const float* __restrict__ Wf1, const float* __restrict__ Wf2,
                                  const float* __restrict__ Wd1, const float* __restrict__ Wd2,
                                  _Float16* __restrict__ Wf1p, _Float16* __restrict__ Wf2p,
                                  _Float16* __restrict__ Wd1p, _Float16* __restrict__ Wd2p) {
  int i = blockIdx.x * 256 + threadIdx.x;
  if (i < NLC * KP * NFC) {              // Wf1 padded K 50->64, fragment-swizzled
    int l = i / (KP * NFC);
    int rem = i % (KP * NFC);
    int k = rem / NFC, n = rem % NFC;
    float v = (k < NGC) ? Wf1[(l * NGC + k) * NFC + n] : 0.f;
    Wf1p[l * KP * NFC + swz_idx(k, n, 2)] = (_Float16)v;
  }
  if (i < NLC * NFC * NFC) {             // K=128 matrices, fragment-swizzled
    int l = i / (NFC * NFC);
    int rem = i % (NFC * NFC);
    int k = rem / NFC, n = rem % NFC;
    int d = l * NFC * NFC + swz_idx(k, n, 4);
    Wf2p[d] = (_Float16)Wf2[i];
    Wd1p[d] = (_Float16)Wd1[i];
    Wd2p[d] = (_Float16)Wd2[i];
  }
}

// ---------------- node encoder ----------------

__global__ void __launch_bounds__(128) enc_pass1(const float* __restrict__ x,
                                                 const float* __restrict__ We1,
                                                 const float* __restrict__ be1,
                                                 float* __restrict__ t,
                                                 float* __restrict__ stats, int N_) {
  __shared__ float sx[64 * XDIM];
  int c = threadIdx.x;
  long long r0 = (long long)blockIdx.x * 64;
  for (int i = c; i < 64 * XDIM; i += 128) {
    long long r = r0 + i / XDIM;
    sx[i] = (r < N_) ? x[r * XDIM + (i % XDIM)] : 0.f;
  }
  __syncthreads();
  float wcol[XDIM];
#pragma unroll
  for (int k = 0; k < XDIM; ++k) wcol[k] = We1[k * HIDC + c];
  float b = be1[c];
  float s = 0.f, sq = 0.f;
  for (int rr = 0; rr < 64; ++rr) {
    long long r = r0 + rr;
    if (r >= N_) break;
    float acc = b;
#pragma unroll
    for (int k = 0; k < XDIM; ++k) acc += sx[rr * XDIM + k] * wcol[k];
    t[r * HIDC + c] = acc;
    s += acc; sq += acc * acc;
  }
  atomicAdd(&stats[c], s);
  atomicAdd(&stats[HIDC + c], sq);
}

__global__ void bn_finalize(float* __restrict__ stats, const float* __restrict__ gamma,
                            const float* __restrict__ beta, int C, float invN) {
  int c = blockIdx.x * blockDim.x + threadIdx.x;
  if (c >= C) return;
  float m = stats[c] * invN;
  float v = stats[C + c] * invN - m * m;
  float rstd = rsqrtf(v + 1e-5f);
  float a = gamma[c] * rstd;
  stats[2 * C + c] = a;
  stats[3 * C + c] = beta[c] - m * a;
}

__global__ void __launch_bounds__(128) enc_pass2(const float* __restrict__ t,
                                                 const float* __restrict__ stats,
                                                 const float* __restrict__ We2,
                                                 const float* __restrict__ be2,
                                                 float* __restrict__ h, int N_) {
  __shared__ float srow[HIDC];
  const float* a  = stats + 2 * HIDC;
  const float* bb = stats + 3 * HIDC;
  int c = threadIdx.x;
  long long r0 = (long long)blockIdx.x * 32;
  float bias = be2[c];
  for (int rr = 0; rr < 32; ++rr) {
    long long r = r0 + rr;
    if (r >= N_) break;
    float tv = t[r * HIDC + c];
    srow[c] = fmaxf(a[c] * tv + bb[c], 0.f);
    __syncthreads();
    float acc = bias;
    for (int k = 0; k < HIDC; ++k) acc += srow[k] * We2[k * HIDC + c];
    h[r * HIDC + c] = acc;
    __syncthreads();
  }
}

// ---------------- gaussian smearing ----------------

__global__ void smear_k(const float* __restrict__ ea3, float* __restrict__ ea,
                        long long total, int E_) {
  long long i = (long long)blockIdx.x * 256 + threadIdx.x;
  if (i >= total) return;
  long long e = i / NGC;
  int k = (int)(i % NGC);
  float dx = ea3[e * 3 + 0], dy = ea3[e * 3 + 1], dz = ea3[e * 3 + 2];
  float d = sqrtf(dx * dx + dy * dy + dz * dz);
  const float step  = CUTOFF / (float)(NGC - 1);
  const float coeff = -0.5f / (step * step);
  float diff = d - (float)k * step;
  ea[i] = expf(coeff * diff * diff);
}

// ---------------- fused edge interaction (WMMA) ----------------
// W = relu(ea@Wf1+bf1)@Wf2+bf2 ; msg = h[row]*W ; atomic agg[col] += msg
__global__ void __launch_bounds__(256) edge_interact(const float* __restrict__ ea,
                                                     const float* __restrict__ hglob,
                                                     const long long* __restrict__ ei,
                                                     const _Float16* __restrict__ W1,
                                                     const _Float16* __restrict__ W2,
                                                     const float* __restrict__ b1,
                                                     const float* __restrict__ b2,
                                                     float* __restrict__ agg, int E_) {
  __shared__ alignas(32) _Float16 sW1[KP * NFC];      // 16 KB (frag-swizzled)
  __shared__ alignas(32) _Float16 sW2[NFC * NFC];     // 32 KB (frag-swizzled)
  __shared__ alignas(32) _Float16 sA [128 * KPP];     // 18 KB (row-major ea tile, padded pitch)
  __shared__ alignas(32) _Float16 sT [128 * HPP];     // 34 KB (row-major intermediate, padded pitch)
  int tid = threadIdx.x;
  // streaming 16B copies of pre-swizzled weights
  for (int i = tid; i < (KP * NFC) / 8;  i += 256) ((uint4*)sW1)[i] = ((const uint4*)W1)[i];
  for (int i = tid; i < (NFC * NFC) / 8; i += 256) ((uint4*)sW2)[i] = ((const uint4*)W2)[i];

  long long e0 = (long long)blockIdx.x * 128;
  // ea tile: 128 edges x 8 runs of 8 halves, one 16B LDS store each
  for (int i = tid; i < 128 * 8; i += 256) {
    int el = i >> 3, k0 = (i & 7) * 8;
    long long e = e0 + el;
    v8h pk;
#pragma unroll
    for (int j = 0; j < 8; ++j) {
      int k = k0 + j;
      float v = (k < NGC && e < E_) ? ea[e * NGC + k] : 0.f;
      pk[j] = (_Float16)v;
    }
    *(v8h*)(sA + el * KPP + k0) = pk;
  }
  __syncthreads();

  int wave = tid >> 5, lane = tid & 31;
  int hf = lane >> 4, lx = lane & 15;
  _Float16* myA = sA + wave * 16 * KPP;
  _Float16* myT = sT + wave * 16 * HPP;

  // GEMM1: [16x64] @ [64x128] -> relu -> f16 stage
  v16h a0 = load_frag_a(myA, KPP, 0,  lane);
  v16h a1 = load_frag_a(myA, KPP, 32, lane);
#pragma unroll
  for (int nt = 0; nt < 8; ++nt) {
    v8f acc = {};
    acc = wmma_f16(a0, load_frag_b_swz(sW1, 2, nt, 0, lane), acc);
    acc = wmma_f16(a1, load_frag_b_swz(sW1, 2, nt, 1, lane), acc);
    float bias = b1[nt * 16 + lx];
#pragma unroll
    for (int i = 0; i < 8; ++i) {
      int m = i + 8 * hf;
      float v = fmaxf(acc[i] + bias, 0.f);
      myT[m * HPP + nt * 16 + lx] = (_Float16)v;     // own-wave region
    }
  }
  // (no block barrier needed: each wave reads only its own sT region)

  // preload per-lane edge indices (each lane touches rows m = 8*hf .. 8*hf+7)
  long long erow[8], ecol[8];
  bool     evld[8];
#pragma unroll
  for (int i = 0; i < 8; ++i) {
    long long e = e0 + wave * 16 + i + 8 * hf;
    evld[i] = (e < E_);
    erow[i] = evld[i] ? ei[e]      : 0;
    ecol[i] = evld[i] ? ei[E_ + e] : 0;
  }

  // GEMM2: [16x128] @ [128x128] ; then msg = h[row]*W, atomic scatter
  v16h A2[4];
#pragma unroll
  for (int kc = 0; kc < 4; ++kc) A2[kc] = load_frag_a(myT, HPP, kc * 32, lane);
#pragma unroll
  for (int nt = 0; nt < 8; ++nt) {
    v8f acc = {};
#pragma unroll
    for (int kc = 0; kc < 4; ++kc)
      acc = wmma_f16(A2[kc], load_frag_b_swz(sW2, 4, nt, kc, lane), acc);
    int n = nt * 16 + lx;
    float bias = b2[n];
#pragma unroll
    for (int i = 0; i < 8; ++i) {
      if (evld[i]) {
        float w   = acc[i] + bias;
        float msg = hglob[erow[i] * HIDC + n] * w;
        atomicAdd(&agg[ecol[i] * HIDC + n], msg);
      }
    }
  }
}

// ---------------- node update (WMMA): h += relu(agg@Wd1+bd1)@Wd2+bd2 ; agg = 0 ----------------

__global__ void __launch_bounds__(256) node_interact(float* __restrict__ hglob,
                                                     float* __restrict__ agg,
                                                     const _Float16* __restrict__ W1,
                                                     const _Float16* __restrict__ W2,
                                                     const float* __restrict__ b1,
                                                     const float* __restrict__ b2, int N_) {
  __shared__ alignas(32) _Float16 sW1[HIDC * HIDC];   // 32 KB (frag-swizzled)
  __shared__ alignas(32) _Float16 sW2[HIDC * HIDC];   // 32 KB (frag-swizzled)
  __shared__ alignas(32) _Float16 sAT[128 * HPP];     // 34 KB (A tile, padded pitch, reused for intermediate)
  int tid = threadIdx.x;
  for (int i = tid; i < (HIDC * HIDC) / 8; i += 256) {
    ((uint4*)sW1)[i] = ((const uint4*)W1)[i];
    ((uint4*)sW2)[i] = ((const uint4*)W2)[i];
  }

  long long n0 = (long long)blockIdx.x * 128;
  // agg tile: 128 rows x 16 runs of 8; float4 loads + zero-store + 16B LDS store
  for (int i = tid; i < 128 * 16; i += 256) {
    int rl = i >> 4, k0 = (i & 15) * 8;
    long long node = n0 + rl;
    v8h pk = {};
    if (node < N_) {
      float* p = agg + node * HIDC + k0;
      float4 f0 = *(const float4*)p;
      float4 f1 = *(const float4*)(p + 4);
      *(float4*)p       = make_float4(0.f, 0.f, 0.f, 0.f);   // reset for next layer
      *(float4*)(p + 4) = make_float4(0.f, 0.f, 0.f, 0.f);
      pk[0] = (_Float16)f0.x; pk[1] = (_Float16)f0.y; pk[2] = (_Float16)f0.z; pk[3] = (_Float16)f0.w;
      pk[4] = (_Float16)f1.x; pk[5] = (_Float16)f1.y; pk[6] = (_Float16)f1.z; pk[7] = (_Float16)f1.w;
    }
    *(v8h*)(sAT + rl * HPP + k0) = pk;
  }
  __syncthreads();

  int wave = tid >> 5, lane = tid & 31;
  int hf = lane >> 4, lx = lane & 15;
  _Float16* myA = sAT + wave * 16 * HPP;

  v16h A[4];
#pragma unroll
  for (int kc = 0; kc < 4; ++kc) A[kc] = load_frag_a(myA, HPP, kc * 32, lane);

#pragma unroll
  for (int nt = 0; nt < 8; ++nt) {
    v8f acc = {};
#pragma unroll
    for (int kc = 0; kc < 4; ++kc)
      acc = wmma_f16(A[kc], load_frag_b_swz(sW1, 4, nt, kc, lane), acc);
    float bias = b1[nt * 16 + lx];
#pragma unroll
    for (int i = 0; i < 8; ++i) {
      int m = i + 8 * hf;
      float v = fmaxf(acc[i] + bias, 0.f);
      myA[m * HPP + nt * 16 + lx] = (_Float16)v;   // own-wave region only
    }
  }

  v16h A2[4];
#pragma unroll
  for (int kc = 0; kc < 4; ++kc) A2[kc] = load_frag_a(myA, HPP, kc * 32, lane);
#pragma unroll
  for (int nt = 0; nt < 8; ++nt) {
    v8f acc = {};
#pragma unroll
    for (int kc = 0; kc < 4; ++kc)
      acc = wmma_f16(A2[kc], load_frag_b_swz(sW2, 4, nt, kc, lane), acc);
    int n = nt * 16 + lx;
    float bias = b2[n];
#pragma unroll
    for (int i = 0; i < 8; ++i) {
      int m = i + 8 * hf;
      long long node = n0 + wave * 16 + m;
      if (node < N_) hglob[node * HIDC + n] += acc[i] + bias;
    }
  }
}

// ---------------- global MLP (G=64 rows; one block) ----------------

__global__ void __launch_bounds__(128) global_mlp(const float* __restrict__ u,
                                                  const float* __restrict__ Wg1, const float* __restrict__ bg1,
                                                  const float* __restrict__ gg1, const float* __restrict__ btg1,
                                                  const float* __restrict__ Wg2, const float* __restrict__ bg2,
                                                  float* __restrict__ up, int G_) {
  __shared__ float su[64 * 16];
  __shared__ float st[64 * HIDC];
  __shared__ float sa[HIDC], sb[HIDC];
  int c = threadIdx.x;
  for (int i = c; i < G_ * 16; i += 128) su[i] = u[i];
  __syncthreads();
  float s = 0.f, sq = 0.f;
  for (int r = 0; r < G_; ++r) {
    float acc = bg1[c];
#pragma unroll
    for (int k = 0; k < 16; ++k) acc += su[r * 16 + k] * Wg1[k * HIDC + c];
    st[r * HIDC + c] = acc;
    s += acc; sq += acc * acc;
  }
  float invG = 1.f / (float)G_;
  float m = s * invG, v = sq * invG - m * m;
  float rstd = rsqrtf(v + 1e-5f);
  float a = gg1[c] * rstd;
  sa[c] = a; sb[c] = btg1[c] - m * a;
  __syncthreads();
  float bias = bg2[c];
  for (int r = 0; r < G_; ++r) {
    float acc = bias;
    for (int k = 0; k < HIDC; ++k)
      acc += fmaxf(sa[k] * st[r * HIDC + k] + sb[k], 0.f) * Wg2[k * HIDC + c];
    up[r * HIDC + c] = acc;
  }
}

// ---------------- shift predictor ----------------

__global__ void __launch_bounds__(64) s1_pass(const float* __restrict__ h, const float* __restrict__ up,
                                              const long long* __restrict__ batch,
                                              const float* __restrict__ Ws1, const float* __restrict__ bs1,
                                              float* __restrict__ z1, float* __restrict__ stats, int N_) {
  __shared__ float srow[2 * HIDC];
  int c = threadIdx.x;
  long long r0 = (long long)blockIdx.x * 32;
  float bias = bs1[c];
  float s = 0.f, sq = 0.f;
  for (int rr = 0; rr < 32; ++rr) {
    long long r = r0 + rr;
    if (r >= N_) break;
    long long g = batch[r];
#pragma unroll
    for (int j = 0; j < 4; ++j) {
      int k = c + 64 * j;
      srow[k] = (k < HIDC) ? h[r * HIDC + k] : up[g * HIDC + (k - HIDC)];
    }
    __syncthreads();
    float acc = bias;
    for (int k = 0; k < 2 * HIDC; ++k) acc += srow[k] * Ws1[k * SHIDC + c];
    z1[r * SHIDC + c] = acc;
    s += acc; sq += acc * acc;
    __syncthreads();
  }
  atomicAdd(&stats[c], s);
  atomicAdd(&stats[SHIDC + c], sq);
}

__global__ void __launch_bounds__(64) s2_pass(const float* __restrict__ z1,
                                              const float* __restrict__ stats1,
                                              const float* __restrict__ Ws2, const float* __restrict__ bs2,
                                              float* __restrict__ z2, float* __restrict__ stats2, int N_) {
  __shared__ float srow[SHIDC];
  const float* a  = stats1 + 2 * SHIDC;
  const float* bb = stats1 + 3 * SHIDC;
  int c = threadIdx.x;
  long long r0 = (long long)blockIdx.x * 64;
  float bias = bs2[c];
  float s = 0.f, sq = 0.f;
  for (int rr = 0; rr < 64; ++rr) {
    long long r = r0 + rr;
    if (r >= N_) break;
    srow[c] = fmaxf(a[c] * z1[r * SHIDC + c] + bb[c], 0.f);
    __syncthreads();
    float acc = bias;
#pragma unroll
    for (int k = 0; k < SHIDC; ++k) acc += srow[k] * Ws2[k * SHIDC + c];
    z2[r * SHIDC + c] = acc;
    s += acc; sq += acc * acc;
    __syncthreads();
  }
  atomicAdd(&stats2[c], s);
  atomicAdd(&stats2[SHIDC + c], sq);
}

__global__ void shifts_pass(const float* __restrict__ z2, const float* __restrict__ stats2,
                            const float* __restrict__ Ws3, const float* __restrict__ bs3,
                            float* __restrict__ out, int N_) {
  long long n = (long long)blockIdx.x * 256 + threadIdx.x;
  if (n >= N_) return;
  const float* a  = stats2 + 2 * SHIDC;
  const float* bb = stats2 + 3 * SHIDC;
  float acc = bs3[0];
#pragma unroll
  for (int k = 0; k < SHIDC; ++k) {
    float v = fmaxf(a[k] * z2[n * SHIDC + k] + bb[k], 0.f);
    acc += v * Ws3[k];
  }
  out[n] = acc;
}

// ---------------- launcher ----------------

extern "C" void kernel_launch(void* const* d_in, const int* in_sizes, int n_in,
                              void* d_out, int out_size, void* d_ws, size_t ws_size,
                              hipStream_t stream) {
  const float*     x    = (const float*)d_in[0];
  const long long* ei   = (const long long*)d_in[1];
  const float*     ea3  = (const float*)d_in[2];
  const long long* batch= (const long long*)d_in[3];
  const float*     u    = (const float*)d_in[4];
  const float* We1 = (const float*)d_in[5];  const float* be1 = (const float*)d_in[6];
  const float* ge1 = (const float*)d_in[7];  const float* bte1= (const float*)d_in[8];
  const float* We2 = (const float*)d_in[9];  const float* be2 = (const float*)d_in[10];
  const float* Wf1 = (const float*)d_in[11]; const float* bf1 = (const float*)d_in[12];
  const float* Wf2 = (const float*)d_in[13]; const float* bf2 = (const float*)d_in[14];
  const float* Wd1 = (const float*)d_in[15]; const float* bd1 = (const float*)d_in[16];
  const float* Wd2 = (const float*)d_in[17]; const float* bd2 = (const float*)d_in[18];
  const float* Wg1 = (const float*)d_in[19]; const float* bg1 = (const float*)d_in[20];
  const float* gg1 = (const float*)d_in[21]; const float* btg1= (const float*)d_in[22];
  const float* Wg2 = (const float*)d_in[23]; const float* bg2 = (const float*)d_in[24];
  const float* Ws1 = (const float*)d_in[25]; const float* bs1 = (const float*)d_in[26];
  const float* gs1 = (const float*)d_in[27]; const float* bts1= (const float*)d_in[28];
  const float* Ws2 = (const float*)d_in[29]; const float* bs2 = (const float*)d_in[30];
  const float* gs2 = (const float*)d_in[31]; const float* bts2= (const float*)d_in[32];
  const float* Ws3 = (const float*)d_in[33]; const float* bs3 = (const float*)d_in[34];

  const int N = in_sizes[0] / XDIM;
  const int E = in_sizes[2] / 3;
  const int G = in_sizes[4] / 16;

  // outputs (concatenated flat: shifts[N], h[N*128], ea[E*50], up[G*128])
  float* out_shifts = (float*)d_out;
  float* out_h  = out_shifts + N;
  float* out_ea = out_h + (long long)N * HIDC;
  float* out_up = out_ea + (long long)E * NGC;

  // workspace layout
  float* t_enc = (float*)d_ws;
  float* agg   = t_enc + (size_t)N * HIDC;
  float* z1    = agg   + (size_t)N * HIDC;
  float* z2    = z1    + (size_t)N * SHIDC;
  float* stats = z2    + (size_t)N * SHIDC;        // 1024 floats
  float* statsEnc = stats;                          // 4*128
  float* statsS1  = stats + 512;                    // 4*64
  float* statsS2  = stats + 768;                    // 4*64
  _Float16* Wf1p = (_Float16*)(stats + 1024);
  _Float16* Wf2p = Wf1p + (size_t)NLC * KP * NFC;
  _Float16* Wd1p = Wf2p + (size_t)NLC * NFC * NFC;
  _Float16* Wd2p = Wd1p + (size_t)NLC * HIDC * HIDC;

  // init: zero agg + BN stat accumulators
  {
    long long nAgg = (long long)N * HIDC;
    int blocks = (int)((nAgg + 255) / 256);
    zero_ws_k<<<blocks, 256, 0, stream>>>(agg, stats, nAgg);
  }
  // convert + fragment-swizzle weights to f16
  convert_weights_k<<<(NLC * NFC * NFC + 255) / 256, 256, 0, stream>>>(
      Wf1, Wf2, Wd1, Wd2, Wf1p, Wf2p, Wd1p, Wd2p);

  // node encoder
  enc_pass1<<<(N + 63) / 64, 128, 0, stream>>>(x, We1, be1, t_enc, statsEnc, N);
  bn_finalize<<<1, 128, 0, stream>>>(statsEnc, ge1, bte1, HIDC, 1.f / (float)N);
  enc_pass2<<<(N + 31) / 32, 128, 0, stream>>>(t_enc, statsEnc, We2, be2, out_h, N);

  // gaussian smearing -> ea output
  {
    long long total = (long long)E * NGC;
    int blocks = (int)((total + 255) / 256);
    smear_k<<<blocks, 256, 0, stream>>>(ea3, out_ea, total, E);
  }

  // 6 interaction layers
  for (int l = 0; l < NLC; ++l) {
    edge_interact<<<(E + 127) / 128, 256, 0, stream>>>(
        out_ea, out_h, ei,
        Wf1p + (size_t)l * KP * NFC, Wf2p + (size_t)l * NFC * NFC,
        bf1 + l * NFC, bf2 + l * NFC, agg, E);
    node_interact<<<(N + 127) / 128, 256, 0, stream>>>(
        out_h, agg,
        Wd1p + (size_t)l * HIDC * HIDC, Wd2p + (size_t)l * HIDC * HIDC,
        bd1 + l * HIDC, bd2 + l * HIDC, N);
  }

  // global MLP
  global_mlp<<<1, 128, 0, stream>>>(u, Wg1, bg1, gg1, btg1, Wg2, bg2, out_up, G);

  // shift predictor
  s1_pass<<<(N + 31) / 32, 64, 0, stream>>>(out_h, out_up, batch, Ws1, bs1, z1, statsS1, N);
  bn_finalize<<<1, 64, 0, stream>>>(statsS1, gs1, bts1, SHIDC, 1.f / (float)N);
  s2_pass<<<(N + 63) / 64, 64, 0, stream>>>(z1, statsS1, Ws2, bs2, z2, statsS2, N);
  bn_finalize<<<1, 64, 0, stream>>>(statsS2, gs2, bts2, SHIDC, 1.f / (float)N);
  shifts_pass<<<(N + 255) / 256, 256, 0, stream>>>(z2, statsS2, Ws3, bs3, out_shifts, N);
}
